// Attention_90675349553279
// MI455X (gfx1250) — matrix operand order repaired and checked by reading, not compiled, not run on hardware
//
#include <hip/hip_runtime.h>
#include <math.h>

// ---------------------------------------------------------------------------
// Sizes for this problem instance
#define BATCH 8
#define SEQ   1024
#define DIM   512
#define HEADS 8
#define DHEAD 64
#define BN    (BATCH * SEQ)          // 8192 rows
#define HB    (HEADS * BATCH)        // 64 attention slabs

// WMMA vector types (gfx1250, wave32)
typedef __bf16 v16bf __attribute__((ext_vector_type(16)));
typedef __bf16 v8bf  __attribute__((ext_vector_type(8)));
typedef float  v8f   __attribute__((ext_vector_type(8)));

// Workspace layout (bytes)
#define OFF_WIN   ((size_t)0)                         // 512*512 bf16 swizzled
#define OFF_WOUT  ((size_t)524288)
#define OFF_F     ((size_t)1048576)                   // f_q, f_k, f_v (h,b,n,d) bf16
#define F_BYTES   ((size_t)8388608)                   // 8*8*1024*64*2
#define OFF_VSWZ  (OFF_F + 3 * F_BYTES)               // f_v in B-fragment layout
#define OFF_NQ    (OFF_VSWZ + F_BYTES)                // 65536 f32 each
#define OFF_MQ    (OFF_NQ + 262144)
#define OFF_NK    (OFF_MQ + 262144)
#define OFF_MK    (OFF_NK + 262144)
#define OFF_CTX   (OFF_MK + 262144)                   // (b,n,h*d) bf16, 8 MB
// total ~42 MB

#define FRAG_BYTES  (32 * 16 * (int)sizeof(__bf16))   // 1 KB per 16x32/32x16 frag
#define FRAG_ELEMS  (32 * 16)

// ---------------------------------------------------------------------------
__device__ __forceinline__ v16bf make_frag(const v8bf lo, const v8bf hi) {
    v16bf f;
#pragma unroll
    for (int e = 0; e < 8; ++e) { f[e] = lo[e]; f[e + 8] = hi[e]; }
    return f;
}
__device__ __forceinline__ v16bf load_frag(const __bf16* p) {
    const v8bf* vp = (const v8bf*)p;
    return make_frag(vp[0], vp[1]);
}

// ---------------------------------------------------------------------------
// Kernel 1: convert + swizzle W_in / W_out (f32 -> bf16 B-fragment layout)
// frag index = kTile(0..15)*32 + nTile(0..31); per frag: 32 lanes x 16 elems
// lane L, elem e  ->  W[kTile*32 + 8*(L>=16) + e + 8*(e>=8)][nTile*16 + (L&15)]
__global__ __launch_bounds__(256) void wswz_kernel(const float* __restrict__ Win,
                                                   const float* __restrict__ Wout,
                                                   __bf16* __restrict__ win_s,
                                                   __bf16* __restrict__ wout_s) {
    const int wave = threadIdx.x >> 5, lane = threadIdx.x & 31;
    const int frag = blockIdx.x * 8 + wave;           // 0..511
    const float* W = blockIdx.y ? Wout : Win;
    __bf16* D     = blockIdx.y ? wout_s : win_s;
    const int kTile = frag >> 5, nTile = frag & 31;
    const int n = nTile * 16 + (lane & 15);
    const int kbase = kTile * 32 + ((lane >= 16) ? 8 : 0);
    __bf16* dst = D + ((size_t)frag * 32 + lane) * 16;
#pragma unroll
    for (int e = 0; e < 16; ++e)
        dst[e] = (__bf16)W[(size_t)(kbase + e + ((e >= 8) ? 8 : 0)) * DIM + n];
}

// ---------------------------------------------------------------------------
// Kernel 2: fused LayerNorm + input projection GEMM (32 rows x 512 per block)
// blockIdx.y selects q / k / v.  8 waves: wave -> (row half, 128-col group),
// 8 f32 accumulators each; per K-step: load 8 B-frags, then 8 wmma.
__global__ __launch_bounds__(256) void proj_kernel(const float* __restrict__ qp,
                                                   const float* __restrict__ kp,
                                                   const float* __restrict__ vp,
                                                   const float* __restrict__ gamma,
                                                   const float* __restrict__ beta,
                                                   const __bf16* __restrict__ Wswz,
                                                   __bf16* __restrict__ f_base) {
    __shared__ __bf16 sA[32][DIM];                    // 32 KB LN output tile

    const int wave = threadIdx.x >> 5, lane = threadIdx.x & 31;
    const int hi = (lane >= 16) ? 1 : 0, lx = lane & 15;
    const float* src = (blockIdx.y == 0) ? qp : (blockIdx.y == 1 ? kp : vp);
    __bf16* fdst = f_base + (size_t)blockIdx.y * (F_BYTES / 2);

    // ---- Phase 1: LayerNorm 4 rows per wave (512 = 32 lanes * 16) ----
#pragma unroll
    for (int rr = 0; rr < 4; ++rr) {
        const int rl = wave * 4 + rr;
        const size_t grow = (size_t)blockIdx.x * 32 + rl;
        const float4* s4 = (const float4*)(src + grow * DIM + lane * 16);
        float xv[16];
        float4 a0 = s4[0], a1 = s4[1], a2 = s4[2], a3 = s4[3];
        xv[0]=a0.x; xv[1]=a0.y; xv[2]=a0.z; xv[3]=a0.w;
        xv[4]=a1.x; xv[5]=a1.y; xv[6]=a1.z; xv[7]=a1.w;
        xv[8]=a2.x; xv[9]=a2.y; xv[10]=a2.z; xv[11]=a2.w;
        xv[12]=a3.x; xv[13]=a3.y; xv[14]=a3.z; xv[15]=a3.w;
        float s = 0.f, s2 = 0.f;
#pragma unroll
        for (int e = 0; e < 16; ++e) { s += xv[e]; s2 += xv[e] * xv[e]; }
#pragma unroll
        for (int off = 16; off >= 1; off >>= 1) {
            s  += __shfl_xor(s,  off, 32);
            s2 += __shfl_xor(s2, off, 32);
        }
        const float mu = s * (1.f / DIM);
        const float var = s2 * (1.f / DIM) - mu * mu;
        const float rs = rsqrtf(var + 1e-5f);
        const float4* g4 = (const float4*)(gamma + lane * 16);
        const float4* b4 = (const float4*)(beta + lane * 16);
        float gv[16], bv[16];
        float4 g0=g4[0],g1=g4[1],g2=g4[2],g3=g4[3];
        float4 b0=b4[0],b1=b4[1],b2=b4[2],b3=b4[3];
        gv[0]=g0.x;gv[1]=g0.y;gv[2]=g0.z;gv[3]=g0.w; gv[4]=g1.x;gv[5]=g1.y;gv[6]=g1.z;gv[7]=g1.w;
        gv[8]=g2.x;gv[9]=g2.y;gv[10]=g2.z;gv[11]=g2.w; gv[12]=g3.x;gv[13]=g3.y;gv[14]=g3.z;gv[15]=g3.w;
        bv[0]=b0.x;bv[1]=b0.y;bv[2]=b0.z;bv[3]=b0.w; bv[4]=b1.x;bv[5]=b1.y;bv[6]=b1.z;bv[7]=b1.w;
        bv[8]=b2.x;bv[9]=b2.y;bv[10]=b2.z;bv[11]=b2.w; bv[12]=b3.x;bv[13]=b3.y;bv[14]=b3.z;bv[15]=b3.w;
#pragma unroll
        for (int e = 0; e < 16; ++e)
            sA[rl][lane * 16 + e] = (__bf16)((xv[e] - mu) * rs * gv[e] + bv[e]);
    }
    __syncthreads();

    // ---- Phase 2: GEMM 32x512 = A(32x512) * W(512x512) on this tile ----
    const int mhalf = wave & 1, ng = wave >> 1;
    const int m0 = mhalf * 16, n0 = ng * 128;
    // per-lane B base pointer (frag = ng*8, kk = 0); constant offsets inside
    const __bf16* bptr = Wswz + ((size_t)(ng * 8) * 32 + lane) * 16;
    const __bf16* aptr = &sA[m0 + lx][hi * 8];

    v8f acc[8] = {};
    for (int kk = 0; kk < 16; ++kk) {
        v16bf af = make_frag(*(const v8bf*)aptr, *(const v8bf*)(aptr + 16));
        v16bf bf[8];
#pragma unroll
        for (int j = 0; j < 8; ++j)
            bf[j] = load_frag(bptr + j * FRAG_ELEMS);
#pragma unroll
        for (int j = 0; j < 8; ++j)
            acc[j] = __builtin_amdgcn_wmma_f32_16x16x32_bf16(
                false, af, false, bf[j], (short)0, acc[j], false, false);
        aptr += 32;                    // next K-chunk in the LN tile row
        bptr += 32 * FRAG_ELEMS;       // next K-chunk: 32 frags forward
    }

    // ---- Store to (h, b, n, d) bf16 ----
#pragma unroll
    for (int j = 0; j < 8; ++j) {
        const int col = n0 + j * 16 + lx;
        const int h = col >> 6, d = col & 63;
#pragma unroll
        for (int r = 0; r < 8; ++r) {
            const size_t grow = (size_t)blockIdx.x * 32 + m0 + r + hi * 8;
            const size_t b = grow >> 10, n = grow & 1023;
            fdst[(((h * (size_t)BATCH + b) * SEQ + n) * DHEAD) + d] = (__bf16)acc[j][r];
        }
    }
}

// ---------------------------------------------------------------------------
// Kernel 3: per-row L2 norm + mean over d=64 for f_q (y=0) and f_k (y=1)
__global__ __launch_bounds__(256) void stats_kernel(const __bf16* __restrict__ f_q,
                                                    const __bf16* __restrict__ f_k,
                                                    float* nq, float* mq,
                                                    float* nk, float* mk) {
    const __bf16* f = blockIdx.y ? f_k : f_q;
    float* np = blockIdx.y ? nk : nq;
    float* mp = blockIdx.y ? mk : mq;
    const int row = blockIdx.x * 256 + threadIdx.x;   // 0..65535
    const __bf16* p = f + (size_t)row * DHEAD;
    float s = 0.f, s2 = 0.f;
#pragma unroll 8
    for (int e = 0; e < DHEAD; ++e) {
        const float x = (float)p[e];
        s += x; s2 += x * x;
    }
    np[row] = sqrtf(s2);
    mp[row] = s * (1.f / DHEAD);
}

// ---------------------------------------------------------------------------
// Kernel 4: swizzle f_v into B-fragment layout (WMMA K axis = key axis)
// fragid = hb*128 + mTile(0..31)*4 + dTile(0..3)
__global__ __launch_bounds__(256) void vswz_kernel(const __bf16* __restrict__ f_v,
                                                   __bf16* __restrict__ vswz) {
    const int wave = threadIdx.x >> 5, lane = threadIdx.x & 31;
    const int fragid = blockIdx.x * 8 + wave;         // 0..8191
    const int hb = fragid >> 7, rem = fragid & 127;
    const int mTile = rem >> 2, dTile = rem & 3;
    const int key = mTile * 32 + ((lane >= 16) ? 16 : 0);
    const int d = dTile * 16 + (lane & 15);
    const __bf16* src = f_v + (size_t)hb * SEQ * DHEAD;
    __bf16* dst = vswz + ((size_t)fragid * 32 + lane) * 16;
#pragma unroll
    for (int e = 0; e < 16; ++e)
        dst[e] = src[(size_t)(key + e) * DHEAD + d];
}

// ---------------------------------------------------------------------------
// Kernel 5: attention.  One (h,b) slab per blockIdx.y; 128 Q rows per block
// (16 per wave).  dots = raw*(alpha/(nq*nk)+beta) - w*mq*mk ; out += dots @ V.
// Per 32-key step: issue K-frag + V-frag loads up front, 4 wmma for raw,
// scale + LDS transpose (hides the V loads), then 4 wmma into the out acc.
__global__ __launch_bounds__(256) void attn_kernel(const __bf16* __restrict__ f_q,
                                                   const __bf16* __restrict__ f_k,
                                                   const __bf16* __restrict__ vswz,
                                                   const float* __restrict__ nq,
                                                   const float* __restrict__ mq,
                                                   const float* __restrict__ nk,
                                                   const float* __restrict__ mk,
                                                   const float* __restrict__ cwp,
                                                   __bf16* __restrict__ ctx) {
    __shared__ __bf16 sT[8][16 * 48];                 // per-wave transpose pad=48

    const int wave = threadIdx.x >> 5, lane = threadIdx.x & 31;
    const int hi = (lane >= 16) ? 1 : 0, lx = lane & 15;
    const int hb = blockIdx.y, h = hb >> 3, b = hb & 7;
    const int q0 = blockIdx.x * 128 + wave * 16;

    const float w = 1.f / (1.f + expf(-cwp[0]));
    const float alpha = 1.f - w, betac = w * (1.f / DHEAD);

    const __bf16* Q = f_q + ((size_t)hb * SEQ + q0) * DHEAD;
    const float* nq_b = nq + hb * SEQ + q0;
    const float* mq_b = mq + hb * SEQ + q0;
    const float* nk_b = nk + hb * SEQ;
    const float* mk_b = mk + hb * SEQ;

    // loop-invariant per-lane Q row stats (VGPR r -> row r + 8*hi)
    float nqr[8], mqr[8];
#pragma unroll
    for (int r = 0; r < 8; ++r) { nqr[r] = nq_b[r + hi * 8]; mqr[r] = mq_b[r + hi * 8]; }

    // loop-invariant Q A-fragments (K = d axis, two chunks of 32)
    v16bf aq[2];
#pragma unroll
    for (int c = 0; c < 2; ++c) {
        const __bf16* p = Q + (size_t)lx * DHEAD + c * 32 + hi * 8;
        aq[c] = make_frag(*(const v8bf*)p, *(const v8bf*)(p + 16));
    }

    // moving per-lane base pointers (constant offsets inside the loop)
    // K frag (tile t, chunk c): kbase + m0*64 + t*16*64 + c*32
    const __bf16* kbase = f_k + (size_t)hb * SEQ * DHEAD + (size_t)lx * DHEAD + hi * 8;
    // V frag (dT):            vbase + (m0/32)*4*FRAG_ELEMS + dT*FRAG_ELEMS
    const __bf16* vbase = vswz + ((size_t)hb * 128 * 32 + lane) * 16;

    v8f oacc[4] = {};
    __bf16* tb = &sT[wave][0];

    for (int m0 = 0; m0 < SEQ; m0 += 32) {
        // ---- issue all global loads for this step up front ----
        v16bf bk[4];                                  // [t*2 + c]
#pragma unroll
        for (int t = 0; t < 2; ++t)
#pragma unroll
            for (int c = 0; c < 2; ++c) {
                const __bf16* p = kbase + t * (16 * DHEAD) + c * 32;
                bk[t * 2 + c] = make_frag(*(const v8bf*)p, *(const v8bf*)(p + 16));
            }
        v16bf bv[4];
#pragma unroll
        for (int dT = 0; dT < 4; ++dT)
            bv[dT] = load_frag(vbase + dT * FRAG_ELEMS);

        // ---- raw = Q K^T for 2 key tiles (16 each), K-dim = 64 ----
        v8f raw0 = {}, raw1 = {};
#pragma unroll
        for (int c = 0; c < 2; ++c) {
            raw0 = __builtin_amdgcn_wmma_f32_16x16x32_bf16(
                false, aq[c], false, bk[c], (short)0, raw0, false, false);
            raw1 = __builtin_amdgcn_wmma_f32_16x16x32_bf16(
                false, aq[c], false, bk[2 + c], (short)0, raw1, false, false);
        }

        // ---- elementwise dots + transpose C-layout -> A-layout via LDS ----
        const float nk0 = nk_b[m0 + lx],      mk0 = mk_b[m0 + lx];
        const float nk1 = nk_b[m0 + 16 + lx], mk1 = mk_b[m0 + 16 + lx];
#pragma unroll
        for (int r = 0; r < 8; ++r) {
            const int qr = r + hi * 8;
            const float inv_nq = 1.f / nqr[r];
            const float d0 = raw0[r] * (alpha * inv_nq / nk0 + betac) - w * mqr[r] * mk0;
            const float d1 = raw1[r] * (alpha * inv_nq / nk1 + betac) - w * mqr[r] * mk1;
            tb[qr * 48 + lx]      = (__bf16)d0;
            tb[qr * 48 + 16 + lx] = (__bf16)d1;
        }
        // same-wave LDS ops are in order; compiler inserts s_wait_dscnt for RAW
        const __bf16* tr = tb + (size_t)lx * 48 + hi * 16;
        v16bf adots = make_frag(*(const v8bf*)tr, *(const v8bf*)(tr + 8));

        // ---- out += dots(16x32) @ V(32x64) ----
#pragma unroll
        for (int dT = 0; dT < 4; ++dT)
            oacc[dT] = __builtin_amdgcn_wmma_f32_16x16x32_bf16(
                false, adots, false, bv[dT], (short)0, oacc[dT], false, false);

        kbase += 32 * DHEAD;          // next 32 keys
        vbase += 4 * FRAG_ELEMS;      // next 4 V frags
    }

    // ---- store ctx (b, n, h*64 + d) bf16 ----
#pragma unroll
    for (int dT = 0; dT < 4; ++dT) {
        const int d = dT * 16 + lx;
#pragma unroll
        for (int r = 0; r < 8; ++r) {
            const size_t n = q0 + r + hi * 8;
            ctx[((size_t)b * SEQ + n) * DIM + h * DHEAD + d] = (__bf16)oacc[dT][r];
        }
    }
}

// ---------------------------------------------------------------------------
// Kernel 6: output projection  out = ctx(8192x512 bf16) @ W_out + b_out  (f32)
__global__ __launch_bounds__(256) void outproj_kernel(const __bf16* __restrict__ ctx,
                                                      const __bf16* __restrict__ Wswz,
                                                      const float* __restrict__ bias,
                                                      float* __restrict__ out) {
    const int wave = threadIdx.x >> 5, lane = threadIdx.x & 31;
    const int hi = (lane >= 16) ? 1 : 0, lx = lane & 15;
    const int mhalf = wave & 1, ng = wave >> 1;
    const size_t m0 = (size_t)blockIdx.x * 32 + mhalf * 16;
    const int n0 = ng * 128;

    const __bf16* aptr = ctx + (m0 + lx) * DIM + hi * 8;
    const __bf16* bptr = Wswz + ((size_t)(ng * 8) * 32 + lane) * 16;

    v8f acc[8] = {};
    for (int kk = 0; kk < 16; ++kk) {
        v16bf af = make_frag(*(const v8bf*)aptr, *(const v8bf*)(aptr + 16));
        v16bf bf[8];
#pragma unroll
        for (int j = 0; j < 8; ++j)
            bf[j] = load_frag(bptr + j * FRAG_ELEMS);
#pragma unroll
        for (int j = 0; j < 8; ++j)
            acc[j] = __builtin_amdgcn_wmma_f32_16x16x32_bf16(
                false, af, false, bf[j], (short)0, acc[j], false, false);
        aptr += 32;
        bptr += 32 * FRAG_ELEMS;
    }
#pragma unroll
    for (int j = 0; j < 8; ++j) {
        const int col = n0 + j * 16 + lx;
        const float bs = bias[col];
#pragma unroll
        for (int r = 0; r < 8; ++r)
            out[(m0 + r + hi * 8) * DIM + col] = acc[j][r] + bs;
    }
}

// ---------------------------------------------------------------------------
extern "C" void kernel_launch(void* const* d_in, const int* in_sizes, int n_in,
                              void* d_out, int out_size, void* d_ws, size_t ws_size,
                              hipStream_t stream) {
    const float* q     = (const float*)d_in[0];
    const float* k     = (const float*)d_in[1];
    const float* v     = (const float*)d_in[2];
    const float* gamma = (const float*)d_in[3];
    const float* beta  = (const float*)d_in[4];
    const float* W_in  = (const float*)d_in[5];
    const float* W_out = (const float*)d_in[6];
    const float* b_out = (const float*)d_in[7];
    const float* cw    = (const float*)d_in[8];
    float* out = (float*)d_out;

    char* ws = (char*)d_ws;
    __bf16* win_s  = (__bf16*)(ws + OFF_WIN);
    __bf16* wout_s = (__bf16*)(ws + OFF_WOUT);
    __bf16* f_base = (__bf16*)(ws + OFF_F);
    __bf16* f_q    = f_base;
    __bf16* f_k    = f_base + (F_BYTES / 2);
    __bf16* f_v    = f_base + 2 * (F_BYTES / 2);
    __bf16* vswz   = (__bf16*)(ws + OFF_VSWZ);
    float*  nq     = (float*)(ws + OFF_NQ);
    float*  mq     = (float*)(ws + OFF_MQ);
    float*  nk     = (float*)(ws + OFF_NK);
    float*  mk     = (float*)(ws + OFF_MK);
    __bf16* ctx    = (__bf16*)(ws + OFF_CTX);

    // 1) weight convert + fragment-swizzle
    wswz_kernel<<<dim3(64, 2), 256, 0, stream>>>(W_in, W_out, win_s, wout_s);
    // 2) fused LN + projection for q, k, v
    proj_kernel<<<dim3(BN / 32, 3), 256, 0, stream>>>(q, k, v, gamma, beta, win_s, f_base);
    // 3) norms & means for q, k
    stats_kernel<<<dim3((HB * SEQ) / 256, 2), 256, 0, stream>>>(f_q, f_k, nq, mq, nk, mk);
    // 4) V fragment-swizzle
    vswz_kernel<<<dim3(1024), 256, 0, stream>>>(f_v, vswz);
    // 5) attention
    attn_kernel<<<dim3(SEQ / 128, HB), 256, 0, stream>>>(f_q, f_k, vswz, nq, mq, nk, mk, cw, ctx);
    // 6) output projection
    outproj_kernel<<<dim3(BN / 32), 256, 0, stream>>>(ctx, wout_s, b_out, out);
}
